// HebbianFW_24807731101744
// MI455X (gfx1250) — compile-verified, matrix operand order checked by reading, not run on hardware
//
#include <hip/hip_runtime.h>
#include <hip/hip_bf16.h>
#include <math.h>

// ---- problem constants (match reference) ----
#define T_STEPS 32
#define BATCH   256
#define OUTD    64
#define IND     256
#define FEATD   128   // 4*4*8 conv features

typedef __attribute__((ext_vector_type(16))) _Float16 v16h;
typedef __attribute__((ext_vector_type(8)))  _Float16 v8h;
typedef __attribute__((ext_vector_type(8)))  float    v8f;

typedef __attribute__((ext_vector_type(4))) unsigned int u32x4;
typedef __attribute__((ext_vector_type(8))) int          i32x8;
typedef __attribute__((ext_vector_type(4))) int          i32x4;

// Combine the two contiguous 8-half chunks of a WMMA A/B fragment.
// Per-lane layout of v_wmma_f32_16x16x32_f16 operands: with kbb = (lane&16)>>1,
// halves 0..7 are K = k0+kbb+0..7 and halves 8..15 are K = k0+16+kbb+0..7.
__device__ __forceinline__ v16h frag_combine(v8h lo, v8h hi) {
    return __builtin_shufflevector(lo, hi, 0, 1, 2, 3, 4, 5, 6, 7,
                                           8, 9, 10, 11, 12, 13, 14, 15);
}

// ---------------------------------------------------------------------------
// Tensor Data Mover: DMA a 2D f16 tile (rows x rowlen, row stride in elems)
// from global memory into LDS at byte offset lds_off. Issued by one wave.
// D# built per CDNA5 ISA (group0 + group1, groups 2/3 zero
// => 2D tensor). data_size=1 (2 bytes), type=2 ("image").
// ---------------------------------------------------------------------------
__device__ __forceinline__ void tdm_load_2d_f16(const _Float16* gsrc,
                                                unsigned lds_off,
                                                unsigned rows, unsigned rowlen,
                                                unsigned row_stride) {
    unsigned long long ga = (unsigned long long)(uintptr_t)gsrc;
    u32x4 g0;
    g0[0] = 1u;                                             // count=1 (valid)
    g0[1] = lds_off;                                        // lds_addr (bytes)
    g0[2] = (unsigned)(ga & 0xFFFFFFFFu);                   // global_addr lo
    g0[3] = (unsigned)((ga >> 32) & 0x01FFFFFFu) | (2u << 30); // addr hi | type=2
    i32x8 g1;
    g1[0] = (int)(1u << 16);                                // data_size = 2B
    g1[1] = (int)((rowlen & 0xFFFFu) << 16);                // tensor_dim0[15:0]
    g1[2] = (int)(((rowlen >> 16) & 0xFFFFu) |
                  ((rows & 0xFFFFu) << 16));                // td0 hi | td1 lo
    g1[3] = (int)(((rows >> 16) & 0xFFFFu) |
                  ((rowlen & 0xFFFFu) << 16));              // td1 hi | tile_dim0
    g1[4] = (int)(rows & 0xFFFFu);                          // tile_dim1 (tile_dim2=0)
    g1[5] = (int)row_stride;                                // tensor_dim0_stride lo
    g1[6] = 0;                                              // stride hi / dim1 stride
    g1[7] = 0;
    i32x4 z4 = {0, 0, 0, 0};
#if __clang_major__ >= 23
    i32x8 z8 = {0, 0, 0, 0, 0, 0, 0, 0};
    __builtin_amdgcn_tensor_load_to_lds(g0, g1, z4, z4, z8, 0);
#else
    __builtin_amdgcn_tensor_load_to_lds(g0, g1, z4, z4, 0);
#endif
}

// ---------------------------------------------------------------------------
// Zero the fast-weight carry
// ---------------------------------------------------------------------------
__global__ void zero_fw_kernel(float* __restrict__ fw) {
    int i = blockIdx.x * blockDim.x + threadIdx.x;   // 64*256 = 16384
    fw[i] = 0.0f;
}

// ---------------------------------------------------------------------------
// Fused conv stack for ONE image: 32x32x3 -> 16x16x8 (relu) -> 8x8x8 (relu)
// -> 4x4x8 (tanh) -> feats_h[128] (f16, feeds the WMMA pipes).
// One 256-thread block per image, all activations staged in LDS.
// SAME padding w/ stride 2 has pad_lo = 0 for all three layers.
// ---------------------------------------------------------------------------
__global__ __launch_bounds__(256)
void conv_feats_kernel(const float* __restrict__ inputs,  // [TB][32][32][3]
                       const float* __restrict__ k1, const float* __restrict__ b1,
                       const float* __restrict__ k2, const float* __restrict__ b2,
                       const float* __restrict__ k3, const float* __restrict__ b3,
                       _Float16* __restrict__ feats_h)     // [TB][128]
{
    __shared__ float s_img[32 * 32 * 3];
    __shared__ float s_c1[16 * 16 * 8];
    __shared__ float s_c2[8 * 8 * 8];
    __shared__ float s_k1[216], s_k2[576], s_k3[576];
    __shared__ float s_b1[8], s_b2[8], s_b3[8];

    const int img = blockIdx.x;
    const int tid = threadIdx.x;

    const float* src = inputs + (size_t)img * (32 * 32 * 3);
    for (int i = tid; i < 3072; i += 256) s_img[i] = src[i];
    for (int i = tid; i < 216;  i += 256) s_k1[i] = k1[i];
    for (int i = tid; i < 576;  i += 256) s_k2[i] = k2[i];
    for (int i = tid; i < 576;  i += 256) s_k3[i] = k3[i];
    if (tid < 8) { s_b1[tid] = b1[tid]; s_b2[tid] = b2[tid]; s_b3[tid] = b3[tid]; }
    __syncthreads();

    // ---- layer 1: 32x32x3 -> 16x16x8
    {
        const int oy = tid >> 4, ox = tid & 15;
        float acc[8];
#pragma unroll
        for (int c = 0; c < 8; ++c) acc[c] = s_b1[c];
#pragma unroll
        for (int ky = 0; ky < 3; ++ky) {
            const int iy = oy * 2 + ky;
            if (iy >= 32) continue;
#pragma unroll
            for (int kx = 0; kx < 3; ++kx) {
                const int ix = ox * 2 + kx;
                if (ix >= 32) continue;
                const float* px = &s_img[(iy * 32 + ix) * 3];
#pragma unroll
                for (int ci = 0; ci < 3; ++ci) {
                    const float v = px[ci];
                    const float* kw = &s_k1[((ky * 3 + kx) * 3 + ci) * 8];
#pragma unroll
                    for (int co = 0; co < 8; ++co) acc[co] = fmaf(v, kw[co], acc[co]);
                }
            }
        }
        float* dst = &s_c1[tid * 8];
#pragma unroll
        for (int co = 0; co < 8; ++co) dst[co] = fmaxf(acc[co], 0.0f);
    }
    __syncthreads();

    // ---- layer 2: 16x16x8 -> 8x8x8
    if (tid < 64) {
        const int oy = tid >> 3, ox = tid & 7;
        float acc[8];
#pragma unroll
        for (int c = 0; c < 8; ++c) acc[c] = s_b2[c];
#pragma unroll
        for (int ky = 0; ky < 3; ++ky) {
            const int iy = oy * 2 + ky;
            if (iy >= 16) continue;
#pragma unroll
            for (int kx = 0; kx < 3; ++kx) {
                const int ix = ox * 2 + kx;
                if (ix >= 16) continue;
                const float* px = &s_c1[(iy * 16 + ix) * 8];
#pragma unroll
                for (int ci = 0; ci < 8; ++ci) {
                    const float v = px[ci];
                    const float* kw = &s_k2[((ky * 3 + kx) * 8 + ci) * 8];
#pragma unroll
                    for (int co = 0; co < 8; ++co) acc[co] = fmaf(v, kw[co], acc[co]);
                }
            }
        }
        float* dst = &s_c2[tid * 8];
#pragma unroll
        for (int co = 0; co < 8; ++co) dst[co] = fmaxf(acc[co], 0.0f);
    }
    __syncthreads();

    // ---- layer 3: 8x8x8 -> 4x4x8, tanh -> f16 feats
    if (tid < 16) {
        const int oy = tid >> 2, ox = tid & 3;
        float acc[8];
#pragma unroll
        for (int c = 0; c < 8; ++c) acc[c] = s_b3[c];
#pragma unroll
        for (int ky = 0; ky < 3; ++ky) {
            const int iy = oy * 2 + ky;
            if (iy >= 8) continue;
#pragma unroll
            for (int kx = 0; kx < 3; ++kx) {
                const int ix = ox * 2 + kx;
                if (ix >= 8) continue;
                const float* px = &s_c2[(iy * 8 + ix) * 8];
#pragma unroll
                for (int ci = 0; ci < 8; ++ci) {
                    const float v = px[ci];
                    const float* kw = &s_k3[((ky * 3 + kx) * 8 + ci) * 8];
#pragma unroll
                    for (int co = 0; co < 8; ++co) acc[co] = fmaf(v, kw[co], acc[co]);
                }
            }
        }
        _Float16* dst = feats_h + (size_t)img * FEATD + tid * 8;
#pragma unroll
        for (int co = 0; co < 8; ++co) dst[co] = (_Float16)tanhf(acc[co]);
    }
}

// ---------------------------------------------------------------------------
// W_eff^T (f16, N-major so WMMA B fragments are K-contiguous b128 loads)
// weffT[n][k] = w[k][n] + coeff[k][n]*fw[k][n]
// ---------------------------------------------------------------------------
__global__ void weff_kernel(const float* __restrict__ w, const float* __restrict__ coeff,
                            const float* __restrict__ fw, _Float16* __restrict__ weffT) {
    int i = blockIdx.x * blockDim.x + threadIdx.x;   // 16384
    int k = i >> 6, n = i & 63;
    weffT[n * IND + k] = (_Float16)(w[i] + coeff[i] * fw[i]);
}

// ---------------------------------------------------------------------------
// Pack x2 = [feats, err, label] into f16, both row-major and transposed.
// ---------------------------------------------------------------------------
__global__ __launch_bounds__(256)
void pack_x2_kernel(const _Float16* __restrict__ feats_t,  // [256][128] f16
                    const float* __restrict__ err,         // [256][64]
                    const float* __restrict__ labels_t,    // [256][64]
                    _Float16* __restrict__ x2h,            // [256][256]
                    _Float16* __restrict__ x2hT)           // [256][256]
{
    const int b = blockIdx.x;      // batch row
    const int k = threadIdx.x;     // feature index
    _Float16 v;
    if (k < FEATD)                 v = feats_t[b * FEATD + k];
    else if (k < FEATD + OUTD)     v = (_Float16)err[b * OUTD + (k - FEATD)];
    else                           v = (_Float16)labels_t[b * OUTD + (k - FEATD - OUTD)];
    x2h[b * IND + k]  = v;
    x2hT[k * IND + b] = v;
}

// ---------------------------------------------------------------------------
// Eval 1: out = feats @ W_eff[:128] + b (aux inputs are zeros -> K=128)
// epilogue: out = tanh(out/5)*5 -> d_out[t];  err = 2*(out-label)/(B*O)
// grid = 16 slabs x 128 threads (4 waves, wave w owns N-tile [16w,16w+16)).
// ---------------------------------------------------------------------------
__global__ __launch_bounds__(128)
void fc_eval1_kernel(const _Float16* __restrict__ feats_t,  // [256][128] f16
                     const _Float16* __restrict__ weffT,    // [64][256] f16
                     const float* __restrict__ lin_b,       // [64]
                     const float* __restrict__ labels_t,    // [256][64]
                     float* __restrict__ out_t,             // [256][64]
                     float* __restrict__ err)               // [256][64]
{
    const int tid  = threadIdx.x;
    const int lane = tid & 31;
    const int wave = tid >> 5;
    const int slab = blockIdx.x;
    const int n0   = wave * 16;
    const int kbb  = (lane & 16) >> 1;           // 0 or 8

    const int arow = slab * 16 + (lane & 15);
    const int bcol = n0 + (lane & 15);
    const _Float16* abase = feats_t + arow * FEATD + kbb;
    const _Float16* bbase = weffT + bcol * IND + kbb;

    v8f acc = {};
#pragma unroll
    for (int k0 = 0; k0 < FEATD; k0 += 32) {
        v16h a = frag_combine(*(const v8h*)(abase + k0), *(const v8h*)(abase + k0 + 16));
        v16h b = frag_combine(*(const v8h*)(bbase + k0), *(const v8h*)(bbase + k0 + 16));
        acc = __builtin_amdgcn_wmma_f32_16x16x32_f16(
            false, a, false, b, (short)0, acc, false, false);
    }

    const int nl = lane & 15;
    const int mb = (lane & 16) >> 1;
#pragma unroll
    for (int r = 0; r < 8; ++r) {
        const int orow = slab * 16 + mb + r;
        const int ocol = n0 + nl;
        const float v = acc[r] + lin_b[ocol];
        const float o = tanhf(v * 0.2f) * 5.0f;               // tanh(v/5)*5
        out_t[orow * OUTD + ocol] = o;
        err[orow * OUTD + ocol] =
            (o - labels_t[orow * OUTD + ocol]) * (2.0f / (float)(BATCH * OUTD));
    }
}

// ---------------------------------------------------------------------------
// Eval 2: logits = x2 @ W_eff + b (K=256) with the A slab DMA'd into LDS by
// the Tensor Data Mover; fused row-softmax -> y2T (f16, transposed for the
// Hebbian-update GEMM).
// ---------------------------------------------------------------------------
__global__ __launch_bounds__(128)
void fc_eval2_kernel(const _Float16* __restrict__ x2h,    // [256][256] f16
                     const _Float16* __restrict__ weffT,  // [64][256] f16
                     const float* __restrict__ lin_b,     // [64]
                     _Float16* __restrict__ y2T)          // [64][256] f16
{
    __shared__ __align__(16) _Float16 s_x[16 * IND];      // 8KB A slab (TDM dest)

    const int tid  = threadIdx.x;
    const int lane = tid & 31;
    const int wave = tid >> 5;
    const int slab = blockIdx.x;
    const int n0   = wave * 16;
    const int kbb  = (lane & 16) >> 1;

    // wave 0 DMAs the 16x256 A slab into LDS, waits, then block syncs
    if (tid < 32) {
        tdm_load_2d_f16(x2h + (size_t)slab * 16 * IND,
                        (unsigned)(uintptr_t)&s_x[0], 16, IND, IND);
        __builtin_amdgcn_s_wait_tensorcnt(0);
    }
    __syncthreads();

    const _Float16* abase = s_x + (lane & 15) * IND + kbb;
    const _Float16* bbase = weffT + (n0 + (lane & 15)) * IND + kbb;

    v8f acc = {};
#pragma unroll
    for (int k0 = 0; k0 < IND; k0 += 32) {
        v16h a = frag_combine(*(const v8h*)(abase + k0), *(const v8h*)(abase + k0 + 16));
        v16h b = frag_combine(*(const v8h*)(bbase + k0), *(const v8h*)(bbase + k0 + 16));
        acc = __builtin_amdgcn_wmma_f32_16x16x32_f16(
            false, a, false, b, (short)0, acc, false, false);
    }

    // reuse the A-slab LDS for the logits (all fragment reads done)
    __syncthreads();
    float* s_log = (float*)s_x;                  // 16x64 f32 = 4KB
    const int nl = lane & 15;
    const int mb = (lane & 16) >> 1;
#pragma unroll
    for (int r = 0; r < 8; ++r) {
        s_log[(mb + r) * OUTD + (n0 + nl)] = acc[r] + lin_b[n0 + nl];
    }
    __syncthreads();

    // row-wise softmax over the 64 columns; one thread per batch row
    if (tid < 16) {
        const int orow = slab * 16 + tid;
        float mx = -1e30f;
        for (int c = 0; c < OUTD; ++c) mx = fmaxf(mx, s_log[tid * OUTD + c]);
        float sum = 0.0f;
        for (int c = 0; c < OUTD; ++c) sum += __expf(s_log[tid * OUTD + c] - mx);
        const float inv = 1.0f / sum;
        for (int c = 0; c < OUTD; ++c)
            y2T[c * IND + orow] = (_Float16)(__expf(s_log[tid * OUTD + c] - mx) * inv);
    }
}

// ---------------------------------------------------------------------------
// Hebbian update: fw = (1-lr)*fw + (lr/B) * x2^T @ y2  (K = batch = 256)
// A = x2hT rows (IN-major, K-contiguous) DMA'd to LDS; B = y2T (N-major).
// ---------------------------------------------------------------------------
__global__ __launch_bounds__(128)
void fw_update_kernel(const _Float16* __restrict__ x2hT,     // [256][256] f16
                      const _Float16* __restrict__ y2T,      // [64][256] f16
                      float* __restrict__ fw,                // [256][64]
                      const float* __restrict__ fw_lr_raw)   // [1]
{
    __shared__ __align__(16) _Float16 s_x[16 * IND];         // 8KB A slab

    const int tid  = threadIdx.x;
    const int lane = tid & 31;
    const int wave = tid >> 5;
    const int slab = blockIdx.x;            // slab of 16 IN-rows
    const int n0   = wave * 16;
    const int kbb  = (lane & 16) >> 1;

    if (tid < 32) {
        tdm_load_2d_f16(x2hT + (size_t)slab * 16 * IND,
                        (unsigned)(uintptr_t)&s_x[0], 16, IND, IND);
        __builtin_amdgcn_s_wait_tensorcnt(0);
    }
    __syncthreads();

    const _Float16* abase = s_x + (lane & 15) * IND + kbb;
    const _Float16* bbase = y2T + (n0 + (lane & 15)) * IND + kbb;

    v8f acc = {};
#pragma unroll
    for (int k0 = 0; k0 < BATCH; k0 += 32) {
        v16h a = frag_combine(*(const v8h*)(abase + k0), *(const v8h*)(abase + k0 + 16));
        v16h b = frag_combine(*(const v8h*)(bbase + k0), *(const v8h*)(bbase + k0 + 16));
        acc = __builtin_amdgcn_wmma_f32_16x16x32_f16(
            false, a, false, b, (short)0, acc, false, false);
    }

    const float lr    = 1.0f / (1.0f + __expf(-fw_lr_raw[0]));   // sigmoid
    const float keep  = 1.0f - lr;
    const float scale = lr * (1.0f / (float)BATCH);

    const int nl = lane & 15;
    const int mb = (lane & 16) >> 1;
#pragma unroll
    for (int r = 0; r < 8; ++r) {
        const int i = slab * 16 + mb + r;
        const int n = n0 + nl;
        fw[i * OUTD + n] = keep * fw[i * OUTD + n] + scale * acc[r];
    }
}

// ---------------------------------------------------------------------------
// host launcher
// ---------------------------------------------------------------------------
extern "C" void kernel_launch(void* const* d_in, const int* in_sizes, int n_in,
                              void* d_out, int out_size, void* d_ws, size_t ws_size,
                              hipStream_t stream) {
    const float* inputs    = (const float*)d_in[0];   // [T,B,32,32,3]
    const float* labels    = (const float*)d_in[1];   // [T,B,64]
    const float* k1        = (const float*)d_in[2];
    const float* b1        = (const float*)d_in[3];
    const float* k2        = (const float*)d_in[4];
    const float* b2        = (const float*)d_in[5];
    const float* k3        = (const float*)d_in[6];
    const float* b3        = (const float*)d_in[7];
    const float* w         = (const float*)d_in[8];
    const float* coeff     = (const float*)d_in[9];
    const float* lin_b     = (const float*)d_in[10];
    const float* fw_lr_raw = (const float*)d_in[11];
    float* out = (float*)d_out;                       // [T,B,64]

    // workspace layout (offsets 256B aligned)
    char* ws = (char*)d_ws;
    float*    fw      = (float*)(ws + 0);             // 16384 f32   (64KB)
    _Float16* feats_h = (_Float16*)(ws + 65536);      // T*B*128 f16 (2MB)
    _Float16* weffT   = (_Float16*)(ws + 2162688);    // 64*256 f16  (32KB)
    float*    err     = (float*)(ws + 2195456);       // 16384 f32   (64KB)
    _Float16* x2h     = (_Float16*)(ws + 2260992);    // 256*256 f16 (128KB)
    _Float16* x2hT    = (_Float16*)(ws + 2392064);    // 256*256 f16 (128KB)
    _Float16* y2T     = (_Float16*)(ws + 2523136);    // 64*256 f16  (32KB)

    // fast-weight carry starts at zero every call (deterministic)
    zero_fw_kernel<<<64, 256, 0, stream>>>(fw);

    // conv stack is independent of the scan carry: batch all T*B images at once
    conv_feats_kernel<<<T_STEPS * BATCH, 256, 0, stream>>>(
        inputs, k1, b1, k2, b2, k3, b3, feats_h);

    // sequential fast-weight scan; stream order enforces the carry dependency
    for (int t = 0; t < T_STEPS; ++t) {
        const _Float16* feats_t = feats_h + (size_t)t * BATCH * FEATD;
        const float* labels_t   = labels  + (size_t)t * BATCH * OUTD;
        float*       out_t      = out     + (size_t)t * BATCH * OUTD;

        weff_kernel<<<64, 256, 0, stream>>>(w, coeff, fw, weffT);
        fc_eval1_kernel<<<16, 128, 0, stream>>>(feats_t, weffT, lin_b, labels_t, out_t, err);
        pack_x2_kernel<<<BATCH, IND, 0, stream>>>(feats_t, err, labels_t, x2h, x2hT);
        fc_eval2_kernel<<<16, 128, 0, stream>>>(x2h, weffT, lin_b, y2T);
        fw_update_kernel<<<16, 128, 0, stream>>>(x2hT, y2T, fw, fw_lr_raw);
    }
}
